// MultiHeadAttentionBlock_72215580115217
// MI455X (gfx1250) — matrix operand order repaired and checked by reading, not compiled, not run on hardware
//
#include <hip/hip_runtime.h>
#include <hip/hip_bf16.h>
#include <stdint.h>

#define B_  2
#define S_  2048
#define D_  1024
#define H_  16
#define DK_ 64
#define BS_ (B_ * S_)   // 4096 rows

typedef __attribute__((ext_vector_type(16))) __bf16 bf16x16;
typedef __attribute__((ext_vector_type(8)))  float  f32x8;
typedef unsigned int v4u __attribute__((ext_vector_type(4)));
typedef unsigned int v8u __attribute__((ext_vector_type(8)));

union FragAB {
    bf16x16 v;
    uint4   q[2];
    unsigned short s[16];
};

__device__ __forceinline__ unsigned short f2bf(float f) {
    unsigned int u = __float_as_uint(f);
    u += 0x7FFFu + ((u >> 16) & 1u);   // round-to-nearest-even
    return (unsigned short)(u >> 16);
}

// ---------------------------------------------------------------------------
// TDM: async-DMA a 2D tile of bf16 (tile_d1 rows x tile_d0 cols) from a
// row-major [tensor_d1, tensor_d0] tensor into LDS (packed row-major).
// D# bitfields per CDNA5 ISA ch.8 (group0: 4 SGPRs, group1: 8 SGPRs).
// One issue per wave; completion tracked with TENSORcnt.
// ---------------------------------------------------------------------------
__device__ __forceinline__ void tdm_load_tile_2d(unsigned int lds_addr,
                                                 const void* gptr,
                                                 unsigned int tensor_d0,
                                                 unsigned int tensor_d1,
                                                 unsigned int tile_d0,
                                                 unsigned int tile_d1)
{
    const unsigned long long ga = (unsigned long long)(uintptr_t)gptr;
    v4u g0;
    g0[0] = 1u;                                        // count=1, user mode
    g0[1] = lds_addr;                                  // LDS byte address
    g0[2] = (unsigned int)ga;                          // global_addr[31:0]
    g0[3] = ((unsigned int)(ga >> 32) & 0x01FFFFFFu)   // global_addr[56:32]
          | 0x80000000u;                               // type=2 ("image")
    v8u g1;
    g1[0] = 1u << 16;                                  // data_size=1 -> 2 bytes
    g1[1] = tensor_d0 << 16;                           // tensor_dim0 lo16 @ bit48
    g1[2] = (tensor_d0 >> 16) | ((tensor_d1 & 0xFFFFu) << 16); // dim0 hi / dim1 lo
    g1[3] = (tensor_d1 >> 16) | (tile_d0 << 16);       // dim1 hi / tile_dim0
    g1[4] = tile_d1 & 0xFFFFu;                         // tile_dim1 (tile_dim2=0)
    g1[5] = tensor_d0;                                 // tensor_dim0_stride lo32
    g1[6] = 0u;                                        // stride hi / dim1_stride
    g1[7] = 0u;
    asm volatile("tensor_load_to_lds %0, %1" :: "s"(g0), "s"(g1) : "memory");
}

// ---------------------------------------------------------------------------
// GEMM: out[row, col] = sum_d X[row, d] * W[col, d]     (x @ W^T)
// Block: 256 thr = 8 waves; tile 128 (rows) x 64 (cols); K-step 32.
// IN_BF16 path stages the X tile with the Tensor Data Mover.
// ---------------------------------------------------------------------------
template <bool IN_BF16, bool OUT_HEADS>
__global__ __launch_bounds__(256)
void gemm_xwt_kernel(const void* __restrict__ Xv,
                     const float* __restrict__ W,
                     void* __restrict__ Outv)
{
    __shared__ __align__(16) unsigned short Xs[128 * 32];  // [row][d]
    __shared__ __align__(16) unsigned short Ws[32 * 64];   // [e][d]

    const int t    = threadIdx.x;
    const int wv   = t >> 5;
    const int lane = t & 31;
    const int lm   = lane & 15;
    const int half = lane >> 4;

    const int rb = blockIdx.x * 128;
    const int cb = blockIdx.y * 64;

    const float*          Xf = (const float*)Xv;
    const unsigned short* Xh = (const unsigned short*)Xv;
    const unsigned int ldsX = (unsigned int)(uintptr_t)&Xs[0];

    f32x8 c[4];
#pragma unroll
    for (int f = 0; f < 4; ++f)
#pragma unroll
        for (int r = 0; r < 8; ++r) c[f][r] = 0.0f;

    for (int k0 = 0; k0 < D_; k0 += 32) {
        // Stage X tile 128x32 bf16 (TDM when already bf16, convert otherwise)
        if (IN_BF16) {
            if (wv == 0)
                tdm_load_tile_2d(ldsX, Xh + (size_t)rb * D_ + k0,
                                 D_, BS_, 32u, 128u);
        } else {
            const int row = t >> 1;
            const int c0  = (t & 1) * 16;
            const size_t src = (size_t)(rb + row) * D_ + k0 + c0;
#pragma unroll
            for (int j = 0; j < 16; ++j) Xs[row * 32 + c0 + j] = f2bf(Xf[src + j]);
        }
        {   // Stage W tile 64x32 as [e][d], bf16
            const int e  = t >> 2;
            const int d0 = (t & 3) * 8;
            const size_t src = (size_t)(cb + e) * D_ + k0 + d0;
#pragma unroll
            for (int j = 0; j < 8; ++j) Ws[e * 32 + d0 + j] = f2bf(W[src + j]);
        }
        if (IN_BF16 && wv == 0) __builtin_amdgcn_s_wait_tensorcnt(0);
        __syncthreads();

        FragAB a;   // rows = wv*16 + lm; K chunks at half*8 / 16+half*8
        {
            const unsigned short* xr = &Xs[(wv * 16 + lm) * 32];
            a.q[0] = *(const uint4*)(xr + half * 8);
            a.q[1] = *(const uint4*)(xr + 16 + half * 8);
        }
#pragma unroll
        for (int f = 0; f < 4; ++f) {
            FragAB b;   // col = f*16 + lm; K chunks at half*8 / 16+half*8
            const unsigned short* wr = &Ws[(f * 16 + lm) * 32];
            b.q[0] = *(const uint4*)(wr + half * 8);
            b.q[1] = *(const uint4*)(wr + 16 + half * 8);
            c[f] = __builtin_amdgcn_wmma_f32_16x16x32_bf16(
                false, a.v, false, b.v, (short)0, c[f], false, false);
        }
        __syncthreads();
    }

#pragma unroll
    for (int f = 0; f < 4; ++f) {
#pragma unroll
        for (int r = 0; r < 8; ++r) {
            const int row = rb + wv * 16 + r + half * 8;
            const int col = cb + f * 16 + lm;
            const float v = c[f][r];
            if (OUT_HEADS) {
                const int b  = row / S_, s = row % S_;
                const int h  = col / DK_, dk = col % DK_;
                ((unsigned short*)Outv)[(((size_t)(b * H_ + h)) * S_ + s) * DK_ + dk] = f2bf(v);
            } else {
                ((float*)Outv)[(size_t)row * D_ + col] = v;
            }
        }
    }
}

// ---------------------------------------------------------------------------
// Flash attention, transposed-score formulation with DOUBLE-BUFFERED TDM:
// wave 0 waits tile j, block barriers, wave 0 issues tile j+1 into the other
// buffer, and all waves run WMMA on tile j while the TDM streams j+1.
// Block = (b,h, 64-query group), 128 thr = 4 waves x 16-query tiles.
// ---------------------------------------------------------------------------
__global__ __launch_bounds__(128)
void attention_kernel(const unsigned short* __restrict__ Qw,
                      const unsigned short* __restrict__ Kw,
                      const unsigned short* __restrict__ Vw,
                      unsigned short* __restrict__ Ow)
{
    __shared__ __align__(16) unsigned short QsT[64 * 64];      // [query][d]
    __shared__ __align__(16) unsigned short Ks[2][32 * 64];    // [key][d] TDM dest
    __shared__ __align__(16) unsigned short Vs[2][32 * 64];    // [key][d] TDM dest
    __shared__ __align__(16) unsigned short Ps[4][16 * 32];    // per-wave P scratch
    __shared__ float RowScale[4][16];                          // alpha / 1/l

    const int t    = threadIdx.x;
    const int wv   = t >> 5;
    const int lane = t & 31;
    const int lm   = lane & 15;
    const int half = lane >> 4;

    const int qg = blockIdx.x;     // 0..31 (query group of 64)
    const int bh = blockIdx.y;     // 0..31 (= b*H + h)
    const int b  = bh >> 4;
    const int h  = bh & 15;

    const int q0b = qg * 64;
    const int q0  = q0b + wv * 16;
    const int qcol = q0 + lm;                  // this lane's query
    const size_t base = (size_t)bh * S_ * DK_;

    const unsigned int ldsK[2] = { (unsigned int)(uintptr_t)&Ks[0][0],
                                   (unsigned int)(uintptr_t)&Ks[1][0] };
    const unsigned int ldsV[2] = { (unsigned int)(uintptr_t)&Vs[0][0],
                                   (unsigned int)(uintptr_t)&Vs[1][0] };

    // Stage Q tile 64x64 as [query][d] (once per block)
    {
        const int row = t >> 1;                // query row within group
        const int c0  = (t & 1) * 32;          // d offset
        const unsigned short* src = Qw + base + (size_t)(q0b + row) * DK_ + c0;
#pragma unroll
        for (int j = 0; j < 32; ++j) QsT[row * 64 + c0 + j] = src[j];
    }

    const int ntile = (q0b + 64) >> 5;         // causal limit for the block
    if (wv == 0) {                             // prologue: prefetch tile 0
        tdm_load_tile_2d(ldsK[0], Kw + base, DK_, S_, DK_, 32u);
        tdm_load_tile_2d(ldsV[0], Vw + base, DK_, S_, DK_, 32u);
    }
    __syncthreads();

    // B-operand fragments of Q^T: col = query lm, K chunks at half*8 / 16+half*8
    FragAB bq[2];
#pragma unroll
    for (int kk = 0; kk < 2; ++kk) {
        const unsigned short* qr = &QsT[(wv * 16 + lm) * 64 + kk * 32];
        bq[kk].q[0] = *(const uint4*)(qr + half * 8);
        bq[kk].q[1] = *(const uint4*)(qr + 16 + half * 8);
    }

    f32x8 o[4];
#pragma unroll
    for (int f = 0; f < 4; ++f)
#pragma unroll
        for (int r = 0; r < 8; ++r) o[f][r] = 0.0f;
    float ms = -3.0e38f, ls = 0.0f;            // per-lane: one query column

    for (int j = 0; j < ntile; ++j) {
        const int k0  = j * 32;
        const int cur = j & 1;
        if (wv == 0) __builtin_amdgcn_s_wait_tensorcnt(0);   // tile j landed
        __syncthreads();                        // all waves: tile j visible,
                                                // tile j-1 compute finished
        if (wv == 0 && (j + 1) < ntile) {       // stream tile j+1 during compute
            const size_t nbase = base + (size_t)(k0 + 32) * DK_;
            tdm_load_tile_2d(ldsK[cur ^ 1], Kw + nbase, DK_, S_, DK_, 32u);
            tdm_load_tile_2d(ldsV[cur ^ 1], Vw + nbase, DK_, S_, DK_, 32u);
        }

        if (k0 <= q0 + 15) {   // wave-uniform
            // S^T frags: sc[g] (g = 16-key group), M=key, N=query
            f32x8 sc[2];
#pragma unroll
            for (int g = 0; g < 2; ++g) {
#pragma unroll
                for (int r = 0; r < 8; ++r) sc[g][r] = 0.0f;
#pragma unroll
                for (int kk = 0; kk < 2; ++kk) {
                    FragAB ak;   // rows = keys g*16+lm, natural [key][d] layout
                    const unsigned short* kr = &Ks[cur][(g * 16 + lm) * 64 + kk * 32];
                    ak.q[0] = *(const uint4*)(kr + half * 8);
                    ak.q[1] = *(const uint4*)(kr + 16 + half * 8);
                    sc[g] = __builtin_amdgcn_wmma_f32_16x16x32_bf16(
                        false, ak.v, false, bq[kk].v, (short)0, sc[g], false, false);
                }
            }
            // Scale + causal mask + register-local row max
            float rm = -3.0e38f;
#pragma unroll
            for (int g = 0; g < 2; ++g)
#pragma unroll
                for (int r = 0; r < 8; ++r) {
                    const int key = k0 + g * 16 + r + half * 8;
                    float sv = sc[g][r] * 0.125f;
                    sv = (key > qcol) ? -3.0e38f : sv;
                    sc[g][r] = sv;
                    rm = fmaxf(rm, sv);
                }
            rm = fmaxf(rm, __shfl_xor(rm, 16, 32));   // merge key-halves
            const float mnew  = fmaxf(ms, rm);
            const float alpha = __expf(ms - mnew);
            float rs = 0.0f;
#pragma unroll
            for (int g = 0; g < 2; ++g)
#pragma unroll
                for (int r = 0; r < 8; ++r) {
                    const float p = __expf(sc[g][r] - mnew);
                    sc[g][r] = p;
                    rs += p;
                }
            rs += __shfl_xor(rs, 16, 32);
            ls = ls * alpha + rs;
            ms = mnew;

            // Distribute alpha to O rows + write P (transposed back) to LDS
            RowScale[wv][lm] = alpha;          // both halves write same value
            unsigned short* pw = &Ps[wv][0];
#pragma unroll
            for (int g = 0; g < 2; ++g)
#pragma unroll
                for (int r = 0; r < 8; ++r)
                    pw[lm * 32 + g * 16 + r + half * 8] = f2bf(sc[g][r]);
            __builtin_amdgcn_wave_barrier();

#pragma unroll
            for (int r = 0; r < 8; ++r) {
                const float al = RowScale[wv][r + half * 8];
#pragma unroll
                for (int f = 0; f < 4; ++f) o[f][r] *= al;
            }
            FragAB ap;   // A = P: rows = queries lm, K chunks over 32 keys
            {
                const unsigned short* pr = &pw[lm * 32];
                ap.q[0] = *(const uint4*)(pr + half * 8);
                ap.q[1] = *(const uint4*)(pr + 16 + half * 8);
            }
#pragma unroll
            for (int f = 0; f < 4; ++f) {
                FragAB bv;   // col = d f*16+lm; keys half*8+i / 16+half*8+i
                const unsigned short* vr = &Vs[cur][f * 16 + lm];
#pragma unroll
                for (int i = 0; i < 8; ++i) {
                    bv.s[i]     = vr[(half * 8 + i) * 64];
                    bv.s[8 + i] = vr[(16 + half * 8 + i) * 64];
                }
                o[f] = __builtin_amdgcn_wmma_f32_16x16x32_bf16(
                    false, ap.v, false, bv.v, (short)0, o[f], false, false);
            }
        }
    }

    // Normalize (distribute 1/l to rows via LDS) and store bf16 [B,S,D]
    RowScale[wv][lm] = 1.0f / ls;
    __builtin_amdgcn_wave_barrier();
#pragma unroll
    for (int r = 0; r < 8; ++r) {
        const float inv = RowScale[wv][r + half * 8];
        const int qrow = q0 + r + half * 8;
        const size_t orow = ((size_t)(b * S_ + qrow)) * D_ + h * DK_;
#pragma unroll
        for (int f = 0; f < 4; ++f)
            Ow[orow + f * 16 + lm] = f2bf(o[f][r] * inv);
    }
}

// ---------------------------------------------------------------------------
extern "C" void kernel_launch(void* const* d_in, const int* in_sizes, int n_in,
                              void* d_out, int out_size, void* d_ws, size_t ws_size,
                              hipStream_t stream)
{
    (void)in_sizes; (void)n_in; (void)out_size; (void)ws_size;

    const float* q  = (const float*)d_in[0];
    const float* k  = (const float*)d_in[1];
    const float* v  = (const float*)d_in[2];
    const float* Wq = (const float*)d_in[3];
    const float* Wk = (const float*)d_in[4];
    const float* Wv = (const float*)d_in[5];
    const float* Wo = (const float*)d_in[6];
    // d_in[7] = causal mask (tril) -- implemented analytically in-kernel.

    const size_t elems = (size_t)BS_ * D_;   // 4M bf16 elems = 8MB each
    unsigned short* Qws = (unsigned short*)d_ws;
    unsigned short* Kws = Qws + elems;
    unsigned short* Vws = Kws + elems;
    unsigned short* Aws = Vws + elems;

    const dim3 gemm_grid(BS_ / 128, D_ / 64);   // 32 x 16
    gemm_xwt_kernel<false, true><<<gemm_grid, 256, 0, stream>>>(q, Wq, Qws);
    gemm_xwt_kernel<false, true><<<gemm_grid, 256, 0, stream>>>(k, Wk, Kws);
    gemm_xwt_kernel<false, true><<<gemm_grid, 256, 0, stream>>>(v, Wv, Vws);

    attention_kernel<<<dim3(S_ / 64, B_ * H_), 128, 0, stream>>>(Qws, Kws, Vws, Aws);

    gemm_xwt_kernel<true, false><<<gemm_grid, 256, 0, stream>>>(Aws, Wo, (float*)d_out);
}
